// SlotAttentionReasoning_68315749810239
// MI455X (gfx1250) — matrix operand order repaired and checked by reading, not compiled, not run on hardware
//
#include <hip/hip_runtime.h>
#include <hip/hip_bf16.h>

#define BDIM     64
#define NTOK     16
#define SLOT_DIM 512
#define HID      512
#define NPROP    64
#define NCLASSES 128

typedef __attribute__((ext_vector_type(16))) __bf16 v16bf;
typedef __attribute__((ext_vector_type(8)))  float  v8f;

// ---------------------------------------------------------------------------
// Kernel 1: per-token MLP. One workgroup (256 threads = 8 waves) per (b,n)
// token. Memory-bound: streams the gathered 1MB w1 slice + 128KB w2 slice.
// Layer 1 uses B128 loads (16B/lane, 512B contiguous per wave per row) to
// minimize VMEM instructions per byte; slot vector is staged into LDS with
// the CDNA5 async-to-LDS DMA path (ASYNCcnt); fp32 FMA keeps exact math.
// ---------------------------------------------------------------------------
__global__ __launch_bounds__(256)
void token_mlp_kernel(const float* __restrict__ slots,
                      const int*   __restrict__ cidxs,
                      const float* __restrict__ fc1_w,
                      const float* __restrict__ fc1_b,
                      const float* __restrict__ fc2_w,
                      const float* __restrict__ fc2_b,
                      float*       __restrict__ p_out) {
  const int token = blockIdx.x;                 // 0 .. B*N-1
  const int c     = cidxs[token];
  const float* __restrict__ w1 = fc1_w + (size_t)c * SLOT_DIM * HID;
  const float* __restrict__ b1 = fc1_b + (size_t)c * HID;
  const float* __restrict__ w2 = fc2_w + (size_t)c * HID * NPROP;
  const float* __restrict__ b2 = fc2_b + (size_t)c * NPROP;
  const float* __restrict__ s  = slots + (size_t)token * SLOT_DIM;

  __shared__ float s_sh[SLOT_DIM];              // slot vector
  __shared__ float hp_sh[2][HID];               // layer-1 partials (2 d-halves)
  __shared__ float h_sh[HID];                   // hidden activations
  __shared__ float p2_sh[8][NPROP];             // layer-2 partials

  const int t = threadIdx.x;                    // 0..255

  // --- stage slot vector (512 floats) via async global->LDS DMA (b64/lane) ---
  {
    unsigned lds_off = (unsigned)(unsigned long long)(const void*)&s_sh[2 * t];
    const float* gp  = s + 2 * t;
    asm volatile("global_load_async_to_lds_b64 %0, %1, off"
                 :: "v"(lds_off), "v"(gp) : "memory");
    asm volatile("s_wait_asynccnt 0" ::: "memory");
  }
  __syncthreads();

  // ---- layer 1: h[k] = relu(sum_d s[d] * w1[d,k] + b1[k]) ----
  // 2 d-halves x 128 column-groups; each thread: 4 columns, 256 rows, B128 loads.
  {
    const int part = t >> 7;                    // d-half: 0 or 1
    const int cg   = (t & 127) * 4;             // column group base
    const int dbeg = part * (SLOT_DIM / 2);
    const int dend = dbeg + (SLOT_DIM / 2);
    float4 acc = make_float4(0.f, 0.f, 0.f, 0.f);
    const float* __restrict__ wp = w1 + cg;
    #pragma unroll 4
    for (int d = dbeg; d < dend; ++d) {
      if ((d & 15) == 0 && d + 32 < dend)       // keep the HBM stream deep
        __builtin_prefetch(wp + (size_t)(d + 32) * HID, 0, 0);
      const float  sd = s_sh[d];
      const float4 w  = *reinterpret_cast<const float4*>(wp + (size_t)d * HID);
      acc.x = fmaf(sd, w.x, acc.x);
      acc.y = fmaf(sd, w.y, acc.y);
      acc.z = fmaf(sd, w.z, acc.z);
      acc.w = fmaf(sd, w.w, acc.w);
    }
    *reinterpret_cast<float4*>(&hp_sh[part][cg]) = acc;
  }
  __syncthreads();

  // combine the two d-halves, add bias, relu
  if (t < 128) {
    const int k = t * 4;
    const float4 p0 = *reinterpret_cast<const float4*>(&hp_sh[0][k]);
    const float4 p1 = *reinterpret_cast<const float4*>(&hp_sh[1][k]);
    const float4 bb = *reinterpret_cast<const float4*>(b1 + k);
    float4 h;
    h.x = fmaxf(p0.x + p1.x + bb.x, 0.f);
    h.y = fmaxf(p0.y + p1.y + bb.y, 0.f);
    h.z = fmaxf(p0.z + p1.z + bb.z, 0.f);
    h.w = fmaxf(p0.w + p1.w + bb.w, 0.f);
    *reinterpret_cast<float4*>(&h_sh[k]) = h;
  }
  __syncthreads();

  // ---- layer 2: p[w] = relu(sum_d h[d] * w2[d,w] + b2[w]) ----
  // 8 d-parts x 32 column-pairs; each thread: 2 columns, 64 rows, B64 loads.
  {
    const int p2 = t >> 5;                      // 0..7
    const int c2 = (t & 31) * 2;                // column pair base
    const int d0 = p2 * (HID / 8);
    float a0 = 0.f, a1 = 0.f;
    #pragma unroll 8
    for (int d = d0; d < d0 + HID / 8; ++d) {
      const float  hv = h_sh[d];
      const float2 w  = *reinterpret_cast<const float2*>(w2 + (size_t)d * NPROP + c2);
      a0 = fmaf(hv, w.x, a0);
      a1 = fmaf(hv, w.y, a1);
    }
    p2_sh[p2][c2]     = a0;
    p2_sh[p2][c2 + 1] = a1;
  }
  __syncthreads();

  if (t < NPROP) {
    float v = b2[t];
    #pragma unroll
    for (int r = 0; r < 8; ++r) v += p2_sh[r][t];
    p_out[(size_t)token * NPROP + t] = fmaxf(v, 0.f);
  }
}

// ---------------------------------------------------------------------------
// Kernel 2: props[b,w] = sum_n p[b,n,w]   (tiny, deterministic reduction)
// ---------------------------------------------------------------------------
__global__ __launch_bounds__(NPROP)
void props_reduce_kernel(const float* __restrict__ p,
                         float*       __restrict__ props) {
  const int b = blockIdx.x;
  const int w = threadIdx.x;
  float acc = 0.f;
  #pragma unroll
  for (int n = 0; n < NTOK; ++n)
    acc += p[((size_t)b * NTOK + n) * NPROP + w];
  props[(size_t)b * NPROP + w] = acc;
}

// ---------------------------------------------------------------------------
// Kernel 3: logits = props @ rw^T + rb via v_wmma_f32_16x16x32_bf16.
// Real GEMM: M=64 (batch), N=128 (classes), K=64 (props). One wave per
// 16x16 output tile, K split into two K=32 WMMAs, f32 accumulation.
// Operand packing follows the CDNA5 wave32 16-bit layouts (ISA 7.12.2):
//   A 16x32 : lanes 0-15 -> M=lane, K in {0-7,16-23}; lanes 16-31 -> {8-15,24-31}
//   B 32x16 : lanes 0-15 -> N=lane, K in {0-7,16-23};          lanes 16-31 -> {8-15,24-31}
//   C/D     : VGPR j -> M = 8*(lane>=16)+j, N = lane&15
// ---------------------------------------------------------------------------
__global__ __launch_bounds__(32)
void classifier_wmma_kernel(const float* __restrict__ props,
                            const float* __restrict__ rw,
                            const float* __restrict__ rb,
                            float*       __restrict__ logits) {
  const int lane = threadIdx.x;       // wave32, EXEC all ones (no divergence)
  const int half = lane >> 4;         // 0: lanes 0-15, 1: lanes 16-31
  const int lr   = lane & 15;
  const int m0   = blockIdx.x * 16;   // batch-row tile base
  const int n0   = blockIdx.y * 16;   // class-col tile base

  v8f acc = {};
  #pragma unroll
  for (int k0 = 0; k0 < NPROP; k0 += 32) {
    v16bf A, Bm;
    // A-matrix: row m0+lr; element i holds K = half*8 + i (i<8)
    //                                  K = half*8 + 16 + (i-8) (i>=8)
    const float* arow = props + (size_t)(m0 + lr) * NPROP + k0 + half * 8;
    #pragma unroll
    for (int i = 0; i < 8; ++i) {
      A[i]     = (__bf16)arow[i];
      A[i + 8] = (__bf16)arow[16 + i];
    }
    // B-matrix: column n0+lr; element i holds K = half*8 + i (i<8)
    //                                     K = half*8 + 16 + (i-8) (i>=8)
    // B[k][n] = rw[n*NPROP + k]  (logits = props @ rw^T)
    const float* brow = rw + (size_t)(n0 + lr) * NPROP + k0 + half * 8;
    #pragma unroll
    for (int i = 0; i < 8; ++i) {
      Bm[i]     = (__bf16)brow[i];
      Bm[i + 8] = (__bf16)brow[16 + i];
    }

    acc = __builtin_amdgcn_wmma_f32_16x16x32_bf16(
        /*neg_a=*/false, A, /*neg_b=*/false, Bm,
        /*c_mod=*/(short)0, acc, /*reuse_a=*/false, /*reuse_b=*/false);
  }

  const float bias = rb[n0 + lr];
  #pragma unroll
  for (int j = 0; j < 8; ++j) {
    const int m = m0 + half * 8 + j;
    logits[(size_t)m * NCLASSES + n0 + lr] = acc[j] + bias;
  }
}

// ---------------------------------------------------------------------------
extern "C" void kernel_launch(void* const* d_in, const int* in_sizes, int n_in,
                              void* d_out, int out_size, void* d_ws, size_t ws_size,
                              hipStream_t stream) {
  const float* slots = (const float*)d_in[0];
  const int*   cidxs = (const int*)  d_in[1];
  const float* fc1_w = (const float*)d_in[2];
  const float* fc1_b = (const float*)d_in[3];
  const float* fc2_w = (const float*)d_in[4];
  const float* fc2_b = (const float*)d_in[5];
  const float* rw    = (const float*)d_in[6];
  const float* rb    = (const float*)d_in[7];
  float* logits = (float*)d_out;

  // scratch layout: p[B*N*NPROP] then props[B*NPROP]  (~272KB total)
  float* p_buf = (float*)d_ws;
  float* props = p_buf + (size_t)BDIM * NTOK * NPROP;

  token_mlp_kernel<<<BDIM * NTOK, 256, 0, stream>>>(slots, cidxs, fc1_w, fc1_b,
                                                    fc2_w, fc2_b, p_buf);
  props_reduce_kernel<<<BDIM, NPROP, 0, stream>>>(p_buf, props);
  classifier_wmma_kernel<<<dim3(BDIM / 16, NCLASSES / 16), 32, 0, stream>>>(
      props, rw, rb, logits);
}
